// ExaoneFlashAttention_59742995087906
// MI455X (gfx1250) — compile-verified, matrix operand order checked by reading, not run on hardware
//
#include <hip/hip_runtime.h>

// ---------------------------------------------------------------------------
// CDNA5 (gfx1250, wave32) attention block:
//   out = FlashAttn(RoPE(X@Wq), RoPE(X@Wk), X@Wv) @ Wo
// Matmuls on v_wmma_f32_16x16x32_bf16; K/V tiles staged to LDS with
// global_load_async_to_lds_b128 (ASYNCcnt) when the toolchain exposes it.
// ---------------------------------------------------------------------------

typedef __attribute__((ext_vector_type(16))) __bf16 v16bf;
typedef __attribute__((ext_vector_type(8)))  float  v8f;

#define DEV __device__ __forceinline__

constexpr int B_  = 4;
constexpr int S_  = 1024;
constexpr int T_  = 4096;          // B*S
constexpr int DM  = 2048;          // d_model
constexpr int NH  = 32;            // query heads
constexpr int NKV = 8;             // kv heads
constexpr int HD  = 64;            // head dim
constexpr float SCALE = 0.125f;    // 64^-0.5

#if __has_builtin(__builtin_amdgcn_global_load_async_to_lds_b128) && \
    __has_builtin(__builtin_amdgcn_s_wait_asynccnt)
#define USE_ASYNC_LDS 1
// exact parameter types per hipcc diagnostic: int4-vector pointers,
// global (AS1, printed "__device__") source and LDS (AS3) destination
typedef int v4i_ __attribute__((vector_size(16)));
typedef __attribute__((address_space(1))) v4i_* as1p_t;
typedef __attribute__((address_space(3))) v4i_* as3p_t;
#endif

DEV v8f wmma_bf16(v16bf a, v16bf b, v8f c) {
  // D = A(16x32) * B(32x16) + C(16x16), f32 accumulate
  return __builtin_amdgcn_wmma_f32_16x16x32_bf16(
      /*neg_a=*/false, a, /*neg_b=*/false, b,
      /*c_mod=*/(short)0, c, /*reuse_a=*/false, /*reuse_b=*/false);
}

// 16x32 bf16 fragment, rows contiguous in memory (row stride = ld elements).
// ISA 16-bit A layout: lanes 0-15 row M=lane, K chunks [0..7],[16..23];
// lanes 16-31 row M=lane-16, K chunks [8..15],[24..31]. Two b128 loads.
DEV v16bf load_frag_row(const __bf16* p, int ld) {
  const int lane = threadIdx.x & 31;
  const __bf16* r = p + (size_t)(lane & 15) * ld + ((lane >> 4) << 3);
  v16bf f;
  reinterpret_cast<uint4*>(&f)[0] = *reinterpret_cast<const uint4*>(r);
  reinterpret_cast<uint4*>(&f)[1] = *reinterpret_cast<const uint4*>(r + 16);
  return f;
}

// Reductions across a 16-lane half of the wave (rows live per-half).
DEV float rmax16(float t) {
#pragma unroll
  for (int o = 1; o <= 8; o <<= 1) t = fmaxf(t, __shfl_xor(t, o, 32));
  return t;
}
DEV float rsum16(float t) {
#pragma unroll
  for (int o = 1; o <= 8; o <<= 1) t += __shfl_xor(t, o, 32);
  return t;
}

// ---------------------------------------------------------------------------
// GEMM: C(MxN) = A(MxK,bf16) * BT(NxK,bf16)^T.  8 waves/WG, 64x64 tile/wave.
// ---------------------------------------------------------------------------
template <typename OutT>
__global__ __launch_bounds__(256) void gemm_bt_wmma(const __bf16* __restrict__ A,
                                                    const __bf16* __restrict__ BT,
                                                    OutT* __restrict__ C,
                                                    int M, int N, int K) {
  const int lane = threadIdx.x & 31;
  const int w    = threadIdx.x >> 5;
  const int m0   = blockIdx.y * 128 + (w >> 2) * 64;
  const int n0   = blockIdx.x * 256 + (w & 3) * 64;

  v8f acc[4][4] = {};
  for (int kk = 0; kk < K; kk += 32) {
    v16bf a[4], b[4];
#pragma unroll
    for (int i = 0; i < 4; ++i)
      a[i] = load_frag_row(A + (size_t)(m0 + 16 * i) * K + kk, K);
#pragma unroll
    for (int j = 0; j < 4; ++j)
      b[j] = load_frag_row(BT + (size_t)(n0 + 16 * j) * K + kk, K);
    if (kk + 32 < K) {   // gfx1250 global_prefetch of next K slab
      __builtin_prefetch(A  + (size_t)m0 * K + kk + 32, 0, 3);
      __builtin_prefetch(BT + (size_t)n0 * K + kk + 32, 0, 3);
    }
#pragma unroll
    for (int i = 0; i < 4; ++i)
#pragma unroll
      for (int j = 0; j < 4; ++j)
        acc[i][j] = wmma_bf16(a[i], b[j], acc[i][j]);
  }

  // C/D layout: VGPR r -> rows (r, r+8) for lane halves; lane&15 -> col
  const int half = lane >> 4, cn = lane & 15;
#pragma unroll
  for (int i = 0; i < 4; ++i)
#pragma unroll
    for (int j = 0; j < 4; ++j)
#pragma unroll
      for (int r = 0; r < 8; ++r) {
        const int row = m0 + 16 * i + r + half * 8;
        const int col = n0 + 16 * j + cn;
        C[(size_t)row * N + col] = (OutT)acc[i][j][r];
      }
}

// ---------------------------------------------------------------------------
// fp32 -> bf16 convert and transposed weight pack (N,K) for B operands
// ---------------------------------------------------------------------------
__global__ void f32_to_bf16_k(const float* __restrict__ x, __bf16* __restrict__ y, int n) {
  const int i = blockIdx.x * blockDim.x + threadIdx.x;
  if (i < n) y[i] = (__bf16)x[i];
}

__global__ void transpose_to_bf16_k(const float* __restrict__ W, __bf16* __restrict__ WT,
                                    int K, int N) {
  const int i = blockIdx.x * blockDim.x + threadIdx.x;
  if (i >= N * K) return;
  const int n = i / K, kk = i - n * K;
  WT[i] = (__bf16)W[(size_t)kk * N + n];
}

// V (T, NKV, 64) -> V^T per head: vt[((b*NKV+h)*64 + d)*S + s]
__global__ void transpose_v_k(const __bf16* __restrict__ v, __bf16* __restrict__ vt) {
  const int i = blockIdx.x * blockDim.x + threadIdx.x;   // B*NKV*HD*S
  const int s  = i & (S_ - 1);
  const int d  = (i >> 10) & (HD - 1);
  const int hb = i >> 16;                // b*NKV + h
  const int h  = hb & (NKV - 1);
  const int b  = hb >> 3;
  vt[i] = v[(size_t)(b * S_ + s) * (NKV * HD) + h * HD + d];
}

// ---------------------------------------------------------------------------
// RoPE (half-rotate form), in place on bf16 (T, H, 64); f32 math
// ---------------------------------------------------------------------------
__global__ void rope_inplace_k(__bf16* __restrict__ x, const float* __restrict__ cs,
                               const float* __restrict__ sn, int H, int total) {
  const int i = blockIdx.x * blockDim.x + threadIdx.x;
  if (i >= total) return;
  const int d  = i & 31;
  const int hh = (i >> 5) % H;
  const int t  = i / (H * 32);
  const size_t base = (size_t)t * H * HD + (size_t)hh * HD + d;
  const float x1 = (float)x[base];
  const float x2 = (float)x[base + 32];
  const float c = cs[t * 32 + d];
  const float s = sn[t * 32 + d];
  x[base]      = (__bf16)(x1 * c - x2 * s);
  x[base + 32] = (__bf16)(x2 * c + x1 * s);
}

// ---------------------------------------------------------------------------
// GQA causal flash attention.
// Grid: (qblk=8, head=32, batch=4); 256 threads = 8 waves, 16 q-rows/wave.
// K (32x64) and V^T (64x32) tiles staged in LDS per 32-key block; async
// LDS-DMA path when available. Online softmax in f32.
// ---------------------------------------------------------------------------
__global__ __launch_bounds__(256) void flash_attn(const __bf16* __restrict__ q,
                                                  const __bf16* __restrict__ k,
                                                  const __bf16* __restrict__ vt,
                                                  __bf16* __restrict__ o) {
  const int qblk = blockIdx.x, h = blockIdx.y, b = blockIdx.z;
  const int kvh  = h >> 2;                       // GROUPS = 4
  const int w    = threadIdx.x >> 5;
  const int lane = threadIdx.x & 31;
  const int half = lane >> 4, cn = lane & 15;
  const int qrow = qblk * 128 + w * 16;          // sequence position base
  const int tok0 = b * S_;

  __shared__ __align__(16) __bf16 Kt[32 * 64];        // 32 keys x 64 dims
  __shared__ __align__(16) __bf16 Vt[64 * 32];        // 64 dims x 32 keys (V^T)
  __shared__ __align__(16) __bf16 Pt[8][16 * 32];

  // Q fragments (16 rows x 64 dims = two 16x32 A-frags), kept in VGPRs
  const __bf16* qp = q + (size_t)(tok0 + qrow) * (NH * HD) + h * HD;
  const v16bf qa0 = load_frag_row(qp, NH * HD);
  const v16bf qa1 = load_frag_row(qp + 32, NH * HD);

  v8f acc[4] = {};                 // 16 q-rows x 64 dims output accumulator
  float mrow[8], lrow[8];
#pragma unroll
  for (int r = 0; r < 8; ++r) { mrow[r] = -1e30f; lrow[r] = 0.f; }

  const int tid = threadIdx.x;
  // K staging: thread -> (key row, 8-elem dim chunk)
  const int krow = tid >> 3, kch = (tid & 7) << 3;
  // V^T staging: thread -> (dim row, 8-elem key chunk)
  const int vrow = tid >> 2, vch = (tid & 3) << 3;
  const __bf16* vhead = vt + (size_t)((b * NKV + kvh) * HD) * S_;

  const int nkb = (qblk + 1) * 4;        // causal: keys [0, q0+128)
  for (int kb = 0; kb < nkb; ++kb) {
    const __bf16* gk = k + (size_t)(tok0 + kb * 32 + krow) * (NKV * HD)
                         + (size_t)kvh * HD + kch;
    const __bf16* gv = vhead + (size_t)vrow * S_ + kb * 32 + vch;
#ifdef USE_ASYNC_LDS
    __builtin_amdgcn_global_load_async_to_lds_b128(
        (as1p_t)(uintptr_t)gk, (as3p_t)&Kt[krow * 64 + kch], 0, 0);
    __builtin_amdgcn_global_load_async_to_lds_b128(
        (as1p_t)(uintptr_t)gv, (as3p_t)&Vt[vrow * 32 + vch], 0, 0);
    __builtin_amdgcn_s_wait_asynccnt(0);
#else
    *reinterpret_cast<uint4*>(&Kt[krow * 64 + kch]) =
        *reinterpret_cast<const uint4*>(gk);
    *reinterpret_cast<uint4*>(&Vt[vrow * 32 + vch]) =
        *reinterpret_cast<const uint4*>(gv);
#endif
    __syncthreads();

    // S = Q @ K^T for 32 keys -> two 16x16 f32 tiles (4 WMMAs)
    v8f s0 = {}, s1 = {};
    {
      v16bf k00 = load_frag_row(Kt, 64);
      v16bf k01 = load_frag_row(Kt + 32, 64);
      s0 = wmma_bf16(qa0, k00, s0);
      s0 = wmma_bf16(qa1, k01, s0);
      v16bf k10 = load_frag_row(Kt + 16 * 64, 64);
      v16bf k11 = load_frag_row(Kt + 16 * 64 + 32, 64);
      s1 = wmma_bf16(qa0, k10, s1);
      s1 = wmma_bf16(qa1, k11, s1);
    }

    // online softmax update per row (rows r / r+8 per lane-half)
    const int c0 = kb * 32 + cn, c1 = c0 + 16;
#pragma unroll
    for (int r = 0; r < 8; ++r) {
      const int row = qrow + r + half * 8;
      const float v0 = (c0 <= row) ? s0[r] * SCALE : -1e30f;
      const float v1 = (c1 <= row) ? s1[r] * SCALE : -1e30f;
      const float mn = fmaxf(mrow[r], rmax16(fmaxf(v0, v1)));
      const float sc = __expf(mrow[r] - mn);
      mrow[r] = mn;
      const float p0 = (c0 <= row) ? __expf(v0 - mn) : 0.f;
      const float p1 = (c1 <= row) ? __expf(v1 - mn) : 0.f;
      lrow[r] = lrow[r] * sc + rsum16(p0 + p1);
#pragma unroll
      for (int j = 0; j < 4; ++j) acc[j][r] *= sc;   // per-row O rescale
      // restage P (16x32) into LDS in row-major for A-fragment reload
      Pt[w][(r + half * 8) * 32 + cn]      = (__bf16)p0;
      Pt[w][(r + half * 8) * 32 + 16 + cn] = (__bf16)p1;
    }

    // O += P(16x32) @ V(32x64); V^T rows are contiguous-key fragments
    const v16bf pa = load_frag_row(Pt[w], 32);
#pragma unroll
    for (int j = 0; j < 4; ++j) {
      v16bf vb = load_frag_row(Vt + 16 * j * 32, 32);
      acc[j] = wmma_bf16(pa, vb, acc[j]);
    }
    __syncthreads();   // protect Kt/Vt before next block's stage
  }

  // normalize and store bf16 (T, 32, 64)
  __bf16* op = o + (size_t)(tok0 + qrow) * (NH * HD) + h * HD;
#pragma unroll
  for (int j = 0; j < 4; ++j)
#pragma unroll
    for (int r = 0; r < 8; ++r) {
      const float val = acc[j][r] / lrow[r];
      op[(size_t)(r + half * 8) * (NH * HD) + j * 16 + cn] = (__bf16)val;
    }
}

// ---------------------------------------------------------------------------
// Launch: X->bf16, pack weights, QKV GEMMs, RoPE, V^T, flash attn, O GEMM
// ---------------------------------------------------------------------------
extern "C" void kernel_launch(void* const* d_in, const int* in_sizes, int n_in,
                              void* d_out, int out_size, void* d_ws, size_t ws_size,
                              hipStream_t stream) {
  (void)in_sizes; (void)n_in; (void)out_size; (void)ws_size;
  const float* hs = (const float*)d_in[0];
  const float* cs = (const float*)d_in[1];
  const float* sn = (const float*)d_in[2];
  const float* Wq = (const float*)d_in[3];
  const float* Wk = (const float*)d_in[4];
  const float* Wv = (const float*)d_in[5];
  const float* Wo = (const float*)d_in[6];
  float* out = (float*)d_out;

  char* ws = (char*)d_ws;
  size_t off = 0;
  auto take = [&](size_t bytes) { char* p = ws + off; off += bytes; return p; };
  __bf16* xb  = (__bf16*)take((size_t)T_ * DM * 2);   // X bf16         16 MiB
  __bf16* wqT = (__bf16*)take((size_t)DM * DM * 2);   // Wq^T bf16       8 MiB
  __bf16* wkT = (__bf16*)take((size_t)512 * DM * 2);  // Wk^T bf16       2 MiB
  __bf16* wvT = (__bf16*)take((size_t)512 * DM * 2);  // Wv^T bf16       2 MiB
  __bf16* woT = (__bf16*)take((size_t)DM * DM * 2);   // Wo^T bf16       8 MiB
  __bf16* qb  = (__bf16*)take((size_t)T_ * DM * 2);   // Q (T,32,64)    16 MiB
  __bf16* kbf = (__bf16*)take((size_t)T_ * 512 * 2);  // K (T,8,64)      4 MiB
  __bf16* vbf = (__bf16*)take((size_t)T_ * 512 * 2);  // V (T,8,64)      4 MiB
  __bf16* vtb = (__bf16*)take((size_t)T_ * 512 * 2);  // V^T per head    4 MiB
  __bf16* ab  = (__bf16*)take((size_t)T_ * DM * 2);   // attn out       16 MiB

  f32_to_bf16_k<<<(T_ * DM) / 256, 256, 0, stream>>>(hs, xb, T_ * DM);
  transpose_to_bf16_k<<<(DM * DM) / 256, 256, 0, stream>>>(Wq, wqT, DM, DM);
  transpose_to_bf16_k<<<(DM * 512) / 256, 256, 0, stream>>>(Wk, wkT, DM, 512);
  transpose_to_bf16_k<<<(DM * 512) / 256, 256, 0, stream>>>(Wv, wvT, DM, 512);
  transpose_to_bf16_k<<<(DM * DM) / 256, 256, 0, stream>>>(Wo, woT, DM, DM);

  const dim3 blk(256);
  gemm_bt_wmma<__bf16><<<dim3(DM / 256, T_ / 128), blk, 0, stream>>>(xb, wqT, qb,  T_, DM,  DM);
  gemm_bt_wmma<__bf16><<<dim3(512 / 256, T_ / 128), blk, 0, stream>>>(xb, wkT, kbf, T_, 512, DM);
  gemm_bt_wmma<__bf16><<<dim3(512 / 256, T_ / 128), blk, 0, stream>>>(xb, wvT, vbf, T_, 512, DM);

  rope_inplace_k<<<(T_ * NH * 32) / 256, 256, 0, stream>>>(qb, cs, sn, NH, T_ * NH * 32);
  rope_inplace_k<<<(T_ * NKV * 32) / 256, 256, 0, stream>>>(kbf, cs, sn, NKV, T_ * NKV * 32);

  transpose_v_k<<<(T_ * 512) / 256, 256, 0, stream>>>(vbf, vtb);

  flash_attn<<<dim3(S_ / 128, NH, B_), blk, 0, stream>>>(qb, kbf, vtb, ab);

  gemm_bt_wmma<float><<<dim3(DM / 256, T_ / 128), blk, 0, stream>>>(ab, woT, out, T_, DM, DM);
}
